// ContrastiveNetWithGAT_22041772163470
// MI455X (gfx1250) — matrix operand order repaired and checked by reading, not compiled, not run on hardware
//
#include <hip/hip_runtime.h>
#include <hip/hip_bf16.h>

typedef __attribute__((ext_vector_type(16))) __bf16 v16bf;
typedef __attribute__((ext_vector_type(8)))  float  v8f;

#define LDSP 34   // padded row pitch (bf16 elems): 68B = 17 banks, coprime with 64

// ---------------------------------------------------------------------------
// bf16 WMMA GEMM:  C[M x Nld] = act( A[M x K] * B[K x Nld] + bias )
// block = 256 threads (8 waves) -> 128 x 128 output tile
// grid  = (ceil(M/128), Nld/128)
// ---------------------------------------------------------------------------
__global__ __launch_bounds__(256)
void gemm_bf16_wmma(const float* __restrict__ A, const float* __restrict__ B,
                    const float* __restrict__ bias, float* __restrict__ C,
                    int M, int K, int Nld, int relu)
{
    __shared__ __bf16 lA[128 * LDSP];
    __shared__ __bf16 lB[128 * LDSP];

    const int tid  = threadIdx.x;
    const int wave = tid >> 5;
    const int lane = tid & 31;
    const int row0 = blockIdx.x * 128;
    const int colB = blockIdx.y * 128;

    v8f acc[8];
#pragma unroll
    for (int ct = 0; ct < 8; ++ct) acc[ct] = (v8f){0.f,0.f,0.f,0.f,0.f,0.f,0.f,0.f};

    for (int k0 = 0; k0 < K; k0 += 32) {
        // stage A tile (rows row0..+127, K k0..+31), f32 -> bf16
        {
            int r  = tid >> 3;            // 0..31
            int c4 = (tid & 7) << 2;      // 0..28 step 4
#pragma unroll
            for (int it = 0; it < 4; ++it) {
                int rr = r + it * 32;
                int gr = row0 + rr;
                float4 v = (gr < M) ? *(const float4*)(A + (size_t)gr * K + k0 + c4)
                                    : make_float4(0.f, 0.f, 0.f, 0.f);
                __bf16* p = lA + rr * LDSP + c4;
                p[0] = (__bf16)v.x; p[1] = (__bf16)v.y;
                p[2] = (__bf16)v.z; p[3] = (__bf16)v.w;
            }
        }
        // stage B tile transposed: lB[n][k] = B[(k0+k)*Nld + colB + n]
        {
#pragma unroll
            for (int it = 0; it < 16; ++it) {
                int idx = it * 256 + tid;
                int k = idx >> 7;         // 0..31
                int n = idx & 127;        // 0..127
                lB[n * LDSP + k] = (__bf16)B[(size_t)(k0 + k) * Nld + colB + n];
            }
        }
        __syncthreads();

        union Frag { v16bf v; unsigned u[8]; };
        // A fragment: 16x32 bf16, interleaved layout (lane<16: K0-7,16-23)
        Frag af;
        {
            const unsigned* pa =
                (const unsigned*)(lA + ((wave << 4) + (lane & 15)) * LDSP);
            const int hi = lane >> 4;
#pragma unroll
            for (int p = 0; p < 8; ++p)
                af.u[p] = pa[(p & 3) + (hi << 2) + ((p >> 2) << 3)];
        }
#pragma unroll
        for (int ct = 0; ct < 8; ++ct) {
            // B fragment: 32x16 bf16, interleaved layout (lane<16: K0-7,16-23)
            Frag bf;
            const unsigned* pb =
                (const unsigned*)(lB + ((ct << 4) + (lane & 15)) * LDSP);
            const int hi = lane >> 4;
#pragma unroll
            for (int p = 0; p < 8; ++p)
                bf.u[p] = pb[(p & 3) + (hi << 2) + ((p >> 2) << 3)];
            acc[ct] = __builtin_amdgcn_wmma_f32_16x16x32_bf16(
                false, af.v, false, bf.v, (short)0, acc[ct], false, false);
        }
        __syncthreads();
    }

    // epilogue: C/D layout VGPR v -> M = v + 8*(lane>=16), N = lane&15
#pragma unroll
    for (int ct = 0; ct < 8; ++ct) {
#pragma unroll
        for (int v = 0; v < 8; ++v) {
            int r = row0 + (wave << 4) + v + ((lane >> 4) << 3);
            int c = colB + (ct << 4) + (lane & 15);
            if (r < M) {
                float val = acc[ct][v];
                if (bias) val += bias[c];
                if (relu) val = fmaxf(val, 0.f);
                C[(size_t)r * Nld + c] = val;
            }
        }
    }
}

// ---------------------------------------------------------------------------
// per-node attention scores: a_src[n,h] = dot(xg[n,h,:], att_src[h,:]) etc.
// thread t = n*8 + j ; j<4 -> src head j, j>=4 -> dst head j-4
// ---------------------------------------------------------------------------
__global__ __launch_bounds__(256)
void att_scores(const float* __restrict__ xg, const float* __restrict__ att_src,
                const float* __restrict__ att_dst, float* __restrict__ a_s,
                float* __restrict__ a_d, int Nn)
{
    int t = blockIdx.x * blockDim.x + threadIdx.x;
    if (t >= Nn * 8) return;
    int j = t & 7, n = t >> 3, h = j & 3;
    const float* row = xg + (size_t)n * 256 + h * 64;
    const float* att = ((j < 4) ? att_src : att_dst) + h * 64;
    float sum = 0.f;
#pragma unroll 8
    for (int c = 0; c < 64; ++c) sum += row[c] * att[c];
    ((j < 4) ? a_s : a_d)[n * 4 + h] = sum;
}

// init: m = -inf, denom = 0, out pre-seeded with bias_g
__global__ __launch_bounds__(256)
void init_bufs(float* __restrict__ m, float* __restrict__ den,
               float* __restrict__ out, const float* __restrict__ bias_g, int Nn)
{
    int i = blockIdx.x * blockDim.x + threadIdx.x;
    if (i < Nn * 256) out[i] = bias_g[i & 255];
    if (i < Nn * 4) { m[i] = -1e30f; den[i] = 0.f; }
}

__device__ __forceinline__ void atomicMaxF(float* addr, float v)
{
    if (v >= 0.f) atomicMax((int*)addr, __float_as_int(v));
    else          atomicMin((unsigned int*)addr, (unsigned int)__float_as_int(v));
}

// pass 1: e = leaky_relu(a_src[src]+a_dst[dst]); segment max into m[dst]
__global__ __launch_bounds__(256)
void edge_logits_max(const int* __restrict__ ei, const float* __restrict__ a_s,
                     const float* __restrict__ a_d, float* __restrict__ ebuf,
                     float* __restrict__ m, int E_, int Nn)
{
    int t = blockIdx.x * blockDim.x + threadIdx.x;
    if (t >= (E_ + Nn) * 4) return;
    int h = t & 3, e = t >> 2;
    int s, d;
    if (e < E_) { s = ei[e]; d = ei[E_ + e]; } else { s = d = e - E_; }
    float v = a_s[s * 4 + h] + a_d[d * 4 + h];
    v = (v > 0.f) ? v : 0.2f * v;
    ebuf[t] = v;
    atomicMaxF(m + d * 4 + h, v);
}

// pass 2: ex = exp(e - m[dst]); segment sum into denom[dst]
__global__ __launch_bounds__(256)
void edge_exp_sum(const int* __restrict__ ei, float* __restrict__ ebuf,
                  const float* __restrict__ m, float* __restrict__ den,
                  int E_, int Nn)
{
    int t = blockIdx.x * blockDim.x + threadIdx.x;
    if (t >= (E_ + Nn) * 4) return;
    int h = t & 3, e = t >> 2;
    int d = (e < E_) ? ei[E_ + e] : (e - E_);
    float ex = __expf(ebuf[t] - m[d * 4 + h]);
    ebuf[t] = ex;
    atomicAdd(den + d * 4 + h, ex);
}

// pass 3: out[dst] += (ex/denom[dst]) * xg[src]; one wave per edge,
// lane -> 8 contiguous channels (head = lane>>3), coalesced 1KB/wave
__global__ __launch_bounds__(256)
void gat_aggregate(const int* __restrict__ ei, const float* __restrict__ ebuf,
                   const float* __restrict__ den, const float* __restrict__ xg,
                   float* __restrict__ out, int E_, int Nn)
{
    int lane = threadIdx.x & 31;
    int e = blockIdx.x * 8 + (threadIdx.x >> 5);
    if (e >= E_ + Nn) return;
    int s, d;
    if (e < E_) { s = ei[e]; d = ei[E_ + e]; } else { s = d = e - E_; }
    int h = lane >> 3;
    float alpha = ebuf[e * 4 + h] / (den[d * 4 + h] + 1e-16f);
    const float4* px = (const float4*)(xg + (size_t)s * 256);
    float4 x0 = px[lane * 2], x1 = px[lane * 2 + 1];
    float* po = out + (size_t)d * 256 + lane * 8;
    atomicAdd(po + 0, alpha * x0.x);
    atomicAdd(po + 1, alpha * x0.y);
    atomicAdd(po + 2, alpha * x0.z);
    atomicAdd(po + 3, alpha * x0.w);
    atomicAdd(po + 4, alpha * x1.x);
    atomicAdd(po + 5, alpha * x1.y);
    atomicAdd(po + 6, alpha * x1.z);
    atomicAdd(po + 7, alpha * x1.w);
}

// ---------------------------------------------------------------------------
extern "C" void kernel_launch(void* const* d_in, const int* in_sizes, int n_in,
                              void* d_out, int out_size, void* d_ws, size_t ws_size,
                              hipStream_t stream)
{
    const float* x        = (const float*)d_in[0];
    const int*   ei       = (const int*)  d_in[1];
    const float* W1       = (const float*)d_in[2];
    const float* b1       = (const float*)d_in[3];
    const float* W2       = (const float*)d_in[4];
    const float* b2       = (const float*)d_in[5];
    const float* Wg       = (const float*)d_in[6];
    const float* att_src  = (const float*)d_in[7];
    const float* att_dst  = (const float*)d_in[8];
    const float* bias_g   = (const float*)d_in[9];

    const int Nn = in_sizes[0] / 512;   // 20000
    const int E_ = in_sizes[1] / 2;     // 320000
    float* out = (float*)d_out;

    // workspace layout (floats)
    float* ws   = (float*)d_ws;
    float* h1   = ws;                               // N*128
    float* h    = h1 + (size_t)Nn * 128;            // N*128
    float* xg   = h  + (size_t)Nn * 128;            // N*256
    float* a_s  = xg + (size_t)Nn * 256;            // N*4
    float* a_d  = a_s + (size_t)Nn * 4;             // N*4
    float* mbuf = a_d + (size_t)Nn * 4;             // N*4
    float* dbuf = mbuf + (size_t)Nn * 4;            // N*4
    float* ebuf = dbuf + (size_t)Nn * 4;            // (E+N)*4

    dim3 blk(256);
    int mBlocks = (Nn + 127) / 128;

    // encoder MLP + GAT projection (bf16 WMMA, f32 accumulate)
    gemm_bf16_wmma<<<dim3(mBlocks, 1), blk, 0, stream>>>(x,  W1, b1, h1, Nn, 512, 128, 1);
    gemm_bf16_wmma<<<dim3(mBlocks, 1), blk, 0, stream>>>(h1, W2, b2, h,  Nn, 128, 128, 0);
    gemm_bf16_wmma<<<dim3(mBlocks, 2), blk, 0, stream>>>(h,  Wg, nullptr, xg, Nn, 128, 256, 0);

    att_scores<<<(Nn * 8 + 255) / 256, blk, 0, stream>>>(xg, att_src, att_dst, a_s, a_d, Nn);
    init_bufs<<<((size_t)Nn * 256 + 255) / 256, blk, 0, stream>>>(mbuf, dbuf, out, bias_g, Nn);

    int T = (E_ + Nn) * 4;
    edge_logits_max<<<(T + 255) / 256, blk, 0, stream>>>(ei, a_s, a_d, ebuf, mbuf, E_, Nn);
    edge_exp_sum  <<<(T + 255) / 256, blk, 0, stream>>>(ei, ebuf, mbuf, dbuf, E_, Nn);
    gat_aggregate <<<(E_ + Nn + 7) / 8, blk, 0, stream>>>(ei, ebuf, dbuf, xg, out, E_, Nn);
}
